// GGNN_65807488909780
// MI455X (gfx1250) — matrix operand order repaired and checked by reading, not compiled, not run on hardware
//
#include <hip/hip_runtime.h>
#include <hip/hip_bf16.h>

// ---------------------------------------------------------------------------
// GGNN forward for MI455X (gfx1250): all GEMMs via v_wmma_f32_16x16x32_bf16,
// f32 accumulate, fused bias + sigmoid/tanh epilogues, on-the-fly f32->bf16
// conversion in the LDS staging stage (so the 184MB adjacency stays f32).
// FAST template path (N%64==0, K%32==0) has branch-free staging.
// ---------------------------------------------------------------------------

typedef __bf16 bf16_t;
typedef __attribute__((ext_vector_type(4)))  __bf16 v4bf;
typedef __attribute__((ext_vector_type(8)))  __bf16 v8bf;
typedef __attribute__((ext_vector_type(16))) __bf16 v16bf;
typedef __attribute__((ext_vector_type(8)))  float  v8f;

static constexpr int Bsz  = 32;
static constexpr int nKG  = 300;
static constexpr int nQ   = 30;
static constexpr int E_   = 8;
static constexpr int H_   = 512;
static constexpr int ANN  = 2;
static constexpr int EMBD = 300;
static constexpr int NANSW = 3000;
static constexpr int NOUTD = 300;
static constexpr int STEPS = 4;

#define BM 128
#define BN 64
#define BK 32

union pkbf { unsigned u; bf16_t h[2]; };

static __device__ inline v8bf cvt8(const float4 a, const float4 b) {
  v8bf r;
  r[0] = (bf16_t)a.x; r[1] = (bf16_t)a.y; r[2] = (bf16_t)a.z; r[3] = (bf16_t)a.w;
  r[4] = (bf16_t)b.x; r[5] = (bf16_t)b.y; r[6] = (bf16_t)b.z; r[7] = (bf16_t)b.w;
  return r;
}

// ---------------------------------------------------------------------------
// Batched WMMA GEMM:  C[z] = act( A[z] @ B[z] + bias[ze] ),  A f32, B f32|bf16
//   z in [0,nz); zb=z/ediv; ze=z%ediv
//   A base += zb*sAb + ze*sAe ; B base += zb*sBb + ze*sBe ; C base += z*sC
// 256 threads / 8 waves; 128x64 block tile; each wave: 2x2 grid of 16x16 tiles
// FAST: requires N%64==0 && K%32==0 (A rows clamped, stores M-guarded).
// ---------------------------------------------------------------------------
template <bool FAST>
__global__ __launch_bounds__(256) void k_wmma_gemm(
    const float* __restrict__ Af, int lda, long long sAb, long long sAe,
    const void* __restrict__ Bp, int ldb, long long sBb, long long sBe, int bF32,
    void* __restrict__ Cp, int ldc, long long sC, int cBf16,
    const float* __restrict__ bias, long long sBias,
    int M, int N, int K, int ediv, int act)
{
  __shared__ __attribute__((aligned(16))) bf16_t As[BM][BK];
  __shared__ __attribute__((aligned(16))) bf16_t Bs[BN][BK + 8];   // n-major, +8 pad

  const int t  = threadIdx.x;
  const int z  = blockIdx.z;
  const int zb = z / ediv, ze = z - zb * ediv;

  const long long offA    = (long long)zb * sAb + (long long)ze * sAe;
  const long long offB    = (long long)zb * sBb + (long long)ze * sBe;
  const long long offC    = (long long)z * sC;
  const long long offBias = (long long)ze * sBias;

  const float*  Bf = (const float*)Bp;
  const bf16_t* Bb = (const bf16_t*)Bp;

  // A staging map: thread -> (row, 16-wide k segment)
  const int sr = t >> 1;
  const int sk = (t & 1) * 16;
  int amr = blockIdx.y * BM + sr;
  if (FAST) amr = amr < (M - 1) ? amr : (M - 1);     // clamp (safe read)
  // B staging map: thread -> (even k pair, 4 n columns)
  const int kp = (t >> 4) * 2;
  const int n0 = (t & 15) * 4;
  const int ngb = blockIdx.x * BN + n0;

  // wave tiling: 8 waves as 4(row)x2(col); each wave 2x2 16x16 tiles
  const int wave = t >> 5, lane = t & 31;
  const int wm = wave >> 1, wn = wave & 1;
  const int lrow = lane & 15, kh = lane >> 4;
  const int ar0 = wm * 32 + lrow, ar1 = ar0 + 16;
  const int bc0 = wn * 32 + lrow, bc1 = bc0 + 16;

  v8f acc00 = {}, acc01 = {}, acc10 = {}, acc11 = {};

  const int ktiles = (K + BK - 1) / BK;
  for (int kt = 0; kt < ktiles; ++kt) {
    const int k0 = kt * BK;
    __syncthreads();
    if (FAST) {
      // ---- A: 4x b128 loads, cvt, 2x ds_store_b128 ----
      const float* ap = Af + offA + (long long)amr * lda + k0 + sk;
      const float4 f0 = *(const float4*)(ap);
      const float4 f1 = *(const float4*)(ap + 4);
      const float4 f2 = *(const float4*)(ap + 8);
      const float4 f3 = *(const float4*)(ap + 12);
      *(v8bf*)&As[sr][sk]     = cvt8(f0, f1);
      *(v8bf*)&As[sr][sk + 8] = cvt8(f2, f3);
      if (kt + 1 < ktiles)
        __builtin_prefetch((const void*)(ap + BK), 0, 0);
      // ---- B: 2 wide loads, pack k-pairs, 4x ds_store_b32 ----
      if (bF32) {
        const float* bp = Bf + offB + (long long)(k0 + kp) * ldb + ngb;
        const float4 r0 = *(const float4*)(bp);
        const float4 r1 = *(const float4*)(bp + ldb);
        const float a0[4] = {r0.x, r0.y, r0.z, r0.w};
        const float a1[4] = {r1.x, r1.y, r1.z, r1.w};
        #pragma unroll
        for (int j = 0; j < 4; ++j) {
          pkbf p; p.h[0] = (bf16_t)a0[j]; p.h[1] = (bf16_t)a1[j];
          *(unsigned*)&Bs[n0 + j][kp] = p.u;
        }
      } else {
        const bf16_t* bp = Bb + offB + (long long)(k0 + kp) * ldb + ngb;
        const v4bf r0 = *(const v4bf*)(bp);
        const v4bf r1 = *(const v4bf*)(bp + ldb);
        #pragma unroll
        for (int j = 0; j < 4; ++j) {
          pkbf p; p.h[0] = r0[j]; p.h[1] = r1[j];
          *(unsigned*)&Bs[n0 + j][kp] = p.u;
        }
      }
    } else {
      // ---- checked staging (edge shapes only) ----
      const int am = blockIdx.y * BM + sr;
      const long long abase = offA + (long long)am * lda;
      #pragma unroll
      for (int j = 0; j < 16; ++j) {
        const int kg = k0 + sk + j;
        float v = 0.0f;
        if (am < M && kg < K) v = Af[abase + kg];
        As[sr][sk + j] = (bf16_t)v;
      }
      #pragma unroll
      for (int j = 0; j < 4; ++j) {
        const int ng = ngb + j;
        float v0 = 0.0f, v1 = 0.0f;
        if (ng < N) {
          if (k0 + kp < K)
            v0 = bF32 ? Bf[offB + (long long)(k0 + kp) * ldb + ng]
                      : (float)Bb[offB + (long long)(k0 + kp) * ldb + ng];
          if (k0 + kp + 1 < K)
            v1 = bF32 ? Bf[offB + (long long)(k0 + kp + 1) * ldb + ng]
                      : (float)Bb[offB + (long long)(k0 + kp + 1) * ldb + ng];
        }
        pkbf p; p.h[0] = (bf16_t)v0; p.h[1] = (bf16_t)v1;
        *(unsigned*)&Bs[n0 + j][kp] = p.u;
      }
    }
    __syncthreads();

    // ---- fragments (ISA 7.12.2): A lanes split K 8+8/16+24; B lanes=col ----
    v8bf a0l = *(const v8bf*)&As[ar0][kh * 8];
    v8bf a0h = *(const v8bf*)&As[ar0][kh * 8 + 16];
    v8bf a1l = *(const v8bf*)&As[ar1][kh * 8];
    v8bf a1h = *(const v8bf*)&As[ar1][kh * 8 + 16];
    v8bf b0l = *(const v8bf*)&Bs[bc0][kh * 8];
    v8bf b0h = *(const v8bf*)&Bs[bc0][kh * 8 + 16];
    v8bf b1l = *(const v8bf*)&Bs[bc1][kh * 8];
    v8bf b1h = *(const v8bf*)&Bs[bc1][kh * 8 + 16];
    v16bf af0, af1, bf0, bf1;
    #pragma unroll
    for (int j = 0; j < 8; ++j) {
      af0[j] = a0l[j]; af0[j + 8] = a0h[j];
      af1[j] = a1l[j]; af1[j + 8] = a1h[j];
      bf0[j] = b0l[j]; bf0[j + 8] = b0h[j];
      bf1[j] = b1l[j]; bf1[j + 8] = b1h[j];
    }
    acc00 = __builtin_amdgcn_wmma_f32_16x16x32_bf16(false, af0, false, bf0, (short)0, acc00, false, false);
    acc01 = __builtin_amdgcn_wmma_f32_16x16x32_bf16(false, af0, false, bf1, (short)0, acc01, false, false);
    acc10 = __builtin_amdgcn_wmma_f32_16x16x32_bf16(false, af1, false, bf0, (short)0, acc10, false, false);
    acc11 = __builtin_amdgcn_wmma_f32_16x16x32_bf16(false, af1, false, bf1, (short)0, acc11, false, false);
  }

  // ---- epilogue: row = v + 8*(lane>=16), col = lane&15, 4 tiles ----
  #pragma unroll
  for (int r = 0; r < 2; ++r) {
    #pragma unroll
    for (int c = 0; c < 2; ++c) {
      v8f acc = r ? (c ? acc11 : acc10) : (c ? acc01 : acc00);
      const int cg = blockIdx.x * BN + wn * 32 + c * 16 + lrow;
      const int mb = blockIdx.y * BM + wm * 32 + r * 16 + kh * 8;
      if (cg < N) {
        const float bv = bias ? bias[offBias + cg] : 0.0f;
        #pragma unroll
        for (int v = 0; v < 8; ++v) {
          const int rg = mb + v;
          if (rg < M) {
            float x = acc[v] + bv;
            if (act == 1)      x = 1.0f / (1.0f + __expf(-x));
            else if (act == 2) x = tanhf(x);
            const long long ci = offC + (long long)rg * ldc + cg;
            if (cBf16) ((bf16_t*)Cp)[ci] = (bf16_t)x;
            else       ((float*)Cp)[ci]  = x;
          }
        }
      }
    }
  }
}

// ---------------------------------------------------------------------------
// Elementwise / prep kernels
// ---------------------------------------------------------------------------
__global__ __launch_bounds__(256) void k_permute_W(
    const float* __restrict__ Ws, const float* __restrict__ bs,
    bf16_t* __restrict__ Wd, float* __restrict__ bd)
{
  long long i = (long long)blockIdx.x * 256 + threadIdx.x;
  const long long total = (long long)E_ * H_ * H_;
  if (i >= total) return;
  const int e = (int)(i / ((long long)H_ * H_));
  const int r = (int)(i % ((long long)H_ * H_));
  const int k = r / H_, h = r % H_;
  Wd[i] = (bf16_t)Ws[(long long)k * (H_ * E_) + h * E_ + e];   // Winp[e][k][h]=Win[k][h*8+e]
  if (k == 0) bd[e * H_ + h] = bs[h * E_ + e];
}

__global__ __launch_bounds__(256) void k_transpose_glove(
    const float* __restrict__ g, float* __restrict__ gT)
{
  long long i = (long long)blockIdx.x * 256 + threadIdx.x;
  const long long total = (long long)NOUTD * NANSW;
  if (i >= total) return;
  const int k = (int)(i / NANSW), a = (int)(i % NANSW);
  gT[i] = g[(long long)a * NOUTD + k];
}

__global__ __launch_bounds__(256) void k_build_xa0(
    const int* __restrict__ idx, const float* __restrict__ anno,
    const float* __restrict__ emb, float* __restrict__ out, long long total)
{
  long long i = (long long)blockIdx.x * 256 + threadIdx.x;
  if (i >= total) return;
  const int c = (int)(i % (EMBD + ANN));
  const long long row = i / (EMBD + ANN);
  float v;
  if (c < EMBD) v = emb[(long long)idx[row] * EMBD + c];
  else          v = anno[row * ANN + (c - EMBD)];
  out[i] = v;
}

__global__ __launch_bounds__(256) void k_build_xa2(
    const float* __restrict__ x, const float* __restrict__ anno,
    float* __restrict__ out, long long total)
{
  long long i = (long long)blockIdx.x * 256 + threadIdx.x;
  if (i >= total) return;
  const int c = (int)(i % (H_ + ANN));
  const long long row = i / (H_ + ANN);
  out[i] = (c < H_) ? x[row * H_ + c] : anno[row * ANN + (c - H_)];
}

__global__ __launch_bounds__(256) void k_copy_x_block(
    float* __restrict__ acat, const float* __restrict__ x, long long total)
{
  long long i = (long long)blockIdx.x * 256 + threadIdx.x;
  if (i >= total) return;
  const long long row = i / H_; const int h = (int)(i % H_);
  acat[row * (3 * H_) + 2 * H_ + h] = x[i];
}

__global__ __launch_bounds__(256) void k_rx_block(
    float* __restrict__ acat, const float* __restrict__ r,
    const float* __restrict__ x, long long total)
{
  long long i = (long long)blockIdx.x * 256 + threadIdx.x;
  if (i >= total) return;
  const long long row = i / H_; const int h = (int)(i % H_);
  acat[row * (3 * H_) + 2 * H_ + h] = r[i] * x[i];
}

__global__ __launch_bounds__(256) void k_gru_update(
    float* __restrict__ x, const float* __restrict__ zg,
    const float* __restrict__ hh, long long total)
{
  long long i = (long long)blockIdx.x * 256 + threadIdx.x;
  if (i >= total) return;
  const float zv = zg[i];
  x[i] = (1.0f - zv) * x[i] + zv * hh[i];
}

__global__ __launch_bounds__(256) void k_agg_reduce(
    const float* __restrict__ ib, const float* __restrict__ jb,
    float* __restrict__ cat, int Nn, int coloff)
{
  int i = blockIdx.x * 256 + threadIdx.x;
  if (i >= Bsz * H_) return;
  const int b = i / H_, h = i % H_;
  float s = 0.0f;
  for (int n = 0; n < Nn; ++n) {
    const long long r = ((long long)b * Nn + n) * H_ + h;
    s += ib[r] * jb[r];
  }
  cat[b * (2 * H_) + coloff + h] = tanhf(s);
}

__global__ __launch_bounds__(256) void k_logsoftmax(float* __restrict__ x, int n)
{
  __shared__ float red[256];
  float* row = x + (long long)blockIdx.x * n;
  const int t = threadIdx.x;
  float m = -1e30f;
  for (int j = t; j < n; j += 256) m = fmaxf(m, row[j]);
  red[t] = m; __syncthreads();
  for (int s = 128; s; s >>= 1) { if (t < s) red[t] = fmaxf(red[t], red[t + s]); __syncthreads(); }
  m = red[0]; __syncthreads();
  float sum = 0.0f;
  for (int j = t; j < n; j += 256) sum += __expf(row[j] - m);
  red[t] = sum; __syncthreads();
  for (int s = 128; s; s >>= 1) { if (t < s) red[t] += red[t + s]; __syncthreads(); }
  const float lse = m + __logf(red[0]);
  __syncthreads();
  for (int j = t; j < n; j += 256) row[j] = row[j] - lse;
}

// ---------------------------------------------------------------------------
// Host-side orchestration
// ---------------------------------------------------------------------------
typedef long long ll;

static inline void gemm(hipStream_t s,
    const float* A, int lda, ll sAb, ll sAe,
    const void* Bp, int ldb, ll sBb, ll sBe, int bF32,
    void* C, int ldc, ll sC, int cBf16,
    const float* bias, ll sBias,
    int M, int N, int K, int nz, int ediv, int act)
{
  dim3 g((N + BN - 1) / BN, (M + BM - 1) / BM, nz);
  if ((N % BN) == 0 && (K % BK) == 0)
    k_wmma_gemm<true><<<g, 256, 0, s>>>(A, lda, sAb, sAe, Bp, ldb, sBb, sBe, bF32,
                                        C, ldc, sC, cBf16, bias, sBias, M, N, K, ediv, act);
  else
    k_wmma_gemm<false><<<g, 256, 0, s>>>(A, lda, sAb, sAe, Bp, ldb, sBb, sBe, bF32,
                                         C, ldc, sC, cBf16, bias, sBias, M, N, K, ediv, act);
}

static inline int grid1(ll n) { return (int)((n + 255) / 256); }

template <class T>
static T* carve(char*& p, size_t n) {
  T* r = (T*)p;
  p += (n * sizeof(T) + 255) & ~(size_t)255;
  return r;
}

static void gp_step(hipStream_t s, float* x, int Nn, const float* adj,
                    const float* Wr, const float* br, const float* Wz, const float* bz,
                    const float* Wt, const float* bt,
                    bf16_t* em, const bf16_t* Winp, const float* binp,
                    const bf16_t* Woutp, const float* boutp,
                    float* acat, float* zb, float* rb, float* hb)
{
  const int NE = Nn * E_;
  const int Mr = Bsz * Nn;
  const ll cnt = (ll)Mr * H_;

  // a_cat[:, 1024:1536] = x
  k_copy_x_block<<<grid1(cnt), 256, 0, s>>>(acat, x, cnt);
  // ins (edge-major bf16, z = b*E+e) then a_in = adj[:, :, :NE] @ ins
  gemm(s, x, H_, (ll)Nn * H_, 0,
       Winp, H_, 0, (ll)H_ * H_, 0,
       em, H_, (ll)Nn * H_, 1,
       binp, H_, Nn, H_, H_, Bsz * E_, E_, 0);
  gemm(s, adj, 2 * NE, (ll)Nn * 2 * NE, 0,
       em, H_, (ll)NE * H_, 0, 0,
       acat, 3 * H_, (ll)Nn * 3 * H_, 0,
       nullptr, 0, Nn, H_, NE, Bsz, 1, 0);
  // outs (reuse em) then a_out = adj[:, :, NE:] @ outs
  gemm(s, x, H_, (ll)Nn * H_, 0,
       Woutp, H_, 0, (ll)H_ * H_, 0,
       em, H_, (ll)Nn * H_, 1,
       boutp, H_, Nn, H_, H_, Bsz * E_, E_, 0);
  gemm(s, adj + NE, 2 * NE, (ll)Nn * 2 * NE, 0,
       em, H_, (ll)NE * H_, 0, 0,
       acat + H_, 3 * H_, (ll)Nn * 3 * H_, 0,
       nullptr, 0, Nn, H_, NE, Bsz, 1, 0);
  // z = sigmoid(a@Wz+bz), r = sigmoid(a@Wr+br)
  gemm(s, acat, 3 * H_, 0, 0, Wz, H_, 0, 0, 1,
       zb, H_, 0, 0, bz, 0, Mr, H_, 3 * H_, 1, 1, 1);
  gemm(s, acat, 3 * H_, 0, 0, Wr, H_, 0, 0, 1,
       rb, H_, 0, 0, br, 0, Mr, H_, 3 * H_, 1, 1, 1);
  // replace third block with r*x, h_hat = tanh(.@Wt+bt)
  k_rx_block<<<grid1(cnt), 256, 0, s>>>(acat, rb, x, cnt);
  gemm(s, acat, 3 * H_, 0, 0, Wt, H_, 0, 0, 1,
       hb, H_, 0, 0, bt, 0, Mr, H_, 3 * H_, 1, 1, 2);
  // x = (1-z)*x + z*h_hat
  k_gru_update<<<grid1(cnt), 256, 0, s>>>(x, zb, hb, cnt);
}

extern "C" void kernel_launch(void* const* d_in, const int* in_sizes, int n_in,
                              void* d_out, int out_size, void* d_ws, size_t ws_size,
                              hipStream_t stream)
{
  (void)in_sizes; (void)n_in; (void)out_size; (void)ws_size;
  // ---- inputs (setup_inputs dict order, params dict in insertion order) ----
  const float* emb   = (const float*)d_in[0];
  const float* Wq    = (const float*)d_in[1];  const float* bq  = (const float*)d_in[2];
  const float* Wk    = (const float*)d_in[3];  const float* bk  = (const float*)d_in[4];
  const float* Win   = (const float*)d_in[5];  const float* bin = (const float*)d_in[6];
  const float* Wout  = (const float*)d_in[7];  const float* bout= (const float*)d_in[8];
  const float* kgWr  = (const float*)d_in[9];  const float* kgbr= (const float*)d_in[10];
  const float* kgWz  = (const float*)d_in[11]; const float* kgbz= (const float*)d_in[12];
  const float* kgWt  = (const float*)d_in[13]; const float* kgbt= (const float*)d_in[14];
  const float* qWr   = (const float*)d_in[15]; const float* qbr = (const float*)d_in[16];
  const float* qWz   = (const float*)d_in[17]; const float* qbz = (const float*)d_in[18];
  const float* qWt   = (const float*)d_in[19]; const float* qbt = (const float*)d_in[20];
  const float* kgWi  = (const float*)d_in[21]; const float* kgbi= (const float*)d_in[22];
  const float* kgWj  = (const float*)d_in[23]; const float* kgbj= (const float*)d_in[24];
  const float* qWi   = (const float*)d_in[25]; const float* qbi = (const float*)d_in[26];
  const float* qWj   = (const float*)d_in[27]; const float* qbj = (const float*)d_in[28];
  const float* Wfc   = (const float*)d_in[29]; const float* bfc = (const float*)d_in[30];
  const float* glove = (const float*)d_in[31];
  const int*   ques  = (const int*)d_in[32];
  const float* adjQ  = (const float*)d_in[33];
  const float* qAnno = (const float*)d_in[34];
  const int*   kgIdx = (const int*)d_in[35];
  const float* adjKG = (const float*)d_in[36];
  const float* kgAnno= (const float*)d_in[37];
  // d_in[38] = n_steps (device scalar; fixed at 4 for determinism)

  // ---- workspace carve ----
  char* w = (char*)d_ws;
  float*  x_kg  = carve<float>(w, (size_t)Bsz * nKG * H_);
  float*  x_q   = carve<float>(w, (size_t)Bsz * nQ  * H_);
  bf16_t* em    = carve<bf16_t>(w, (size_t)Bsz * nKG * E_ * H_);  // ins/outs, reused
  float*  acat  = carve<float>(w, (size_t)Bsz * nKG * 3 * H_);
  float*  zbuf  = carve<float>(w, (size_t)Bsz * nKG * H_);
  float*  rbuf  = carve<float>(w, (size_t)Bsz * nKG * H_);
  float*  hbuf  = carve<float>(w, (size_t)Bsz * nKG * H_);
  float*  xa    = carve<float>(w, (size_t)Bsz * nKG * (H_ + ANN));
  bf16_t* Winp  = carve<bf16_t>(w, (size_t)E_ * H_ * H_);
  float*  binp  = carve<float>(w, (size_t)E_ * H_);
  bf16_t* Woutp = carve<bf16_t>(w, (size_t)E_ * H_ * H_);
  float*  boutp = carve<float>(w, (size_t)E_ * H_);
  float*  gT    = carve<float>(w, (size_t)NOUTD * NANSW);
  float*  cat   = carve<float>(w, (size_t)Bsz * 2 * H_);
  float*  fcb   = carve<float>(w, (size_t)Bsz * NOUTD);

  // ---- prep: permuted edge weights (bf16) + glove^T ----
  k_permute_W<<<grid1((ll)E_ * H_ * H_), 256, 0, stream>>>(Win, bin, Winp, binp);
  k_permute_W<<<grid1((ll)E_ * H_ * H_), 256, 0, stream>>>(Wout, bout, Woutp, boutp);
  k_transpose_glove<<<grid1((ll)NOUTD * NANSW), 256, 0, stream>>>(glove, gT);

  // ---- initial projections: x = [emb[idx], anno] @ W{k,q} + b ----
  {
    const ll tot = (ll)Bsz * nKG * (EMBD + ANN);
    k_build_xa0<<<grid1(tot), 256, 0, stream>>>(kgIdx, kgAnno, emb, xa, tot);
    gemm(stream, xa, EMBD + ANN, 0, 0, Wk, H_, 0, 0, 1,
         x_kg, H_, 0, 0, bk, 0, Bsz * nKG, H_, EMBD + ANN, 1, 1, 0);
  }
  {
    const ll tot = (ll)Bsz * nQ * (EMBD + ANN);
    k_build_xa0<<<grid1(tot), 256, 0, stream>>>(ques, qAnno, emb, xa, tot);
    gemm(stream, xa, EMBD + ANN, 0, 0, Wq, H_, 0, 0, 1,
         x_q, H_, 0, 0, bq, 0, Bsz * nQ, H_, EMBD + ANN, 1, 1, 0);
  }

  // ---- propagation ----
  for (int s = 0; s < STEPS; ++s)
    gp_step(stream, x_kg, nKG, adjKG, kgWr, kgbr, kgWz, kgbz, kgWt, kgbt,
            em, Winp, binp, Woutp, boutp, acat, zbuf, rbuf, hbuf);
  for (int s = 0; s < STEPS; ++s)
    gp_step(stream, x_q, nQ, adjQ, qWr, qbr, qWz, qbz, qWt, qbt,
            em, Winp, binp, Woutp, boutp, acat, zbuf, rbuf, hbuf);

  // ---- aggregation: tanh( sum_n sigmoid(xa@Wi+bi) * tanh(xa@Wj+bj) ) ----
  {
    const ll tot = (ll)Bsz * nKG * (H_ + ANN);
    k_build_xa2<<<grid1(tot), 256, 0, stream>>>(x_kg, kgAnno, xa, tot);
    gemm(stream, xa, H_ + ANN, 0, 0, kgWi, H_, 0, 0, 1,
         zbuf, H_, 0, 0, kgbi, 0, Bsz * nKG, H_, H_ + ANN, 1, 1, 1);
    gemm(stream, xa, H_ + ANN, 0, 0, kgWj, H_, 0, 0, 1,
         rbuf, H_, 0, 0, kgbj, 0, Bsz * nKG, H_, H_ + ANN, 1, 1, 2);
    k_agg_reduce<<<grid1(Bsz * H_), 256, 0, stream>>>(zbuf, rbuf, cat, nKG, 0);
  }
  {
    const ll tot = (ll)Bsz * nQ * (H_ + ANN);
    k_build_xa2<<<grid1(tot), 256, 0, stream>>>(x_q, qAnno, xa, tot);
    gemm(stream, xa, H_ + ANN, 0, 0, qWi, H_, 0, 0, 1,
         zbuf, H_, 0, 0, qbi, 0, Bsz * nQ, H_, H_ + ANN, 1, 1, 1);
    gemm(stream, xa, H_ + ANN, 0, 0, qWj, H_, 0, 0, 1,
         rbuf, H_, 0, 0, qbj, 0, Bsz * nQ, H_, H_ + ANN, 1, 1, 2);
    k_agg_reduce<<<grid1(Bsz * H_), 256, 0, stream>>>(zbuf, rbuf, cat, nQ, H_);
  }

  // ---- head: fc = cat@Wfc+bfc ; logits = fc@glove^T ; log_softmax ----
  gemm(stream, cat, 2 * H_, 0, 0, Wfc, NOUTD, 0, 0, 1,
       fcb, NOUTD, 0, 0, bfc, 0, Bsz, NOUTD, 2 * H_, 1, 1, 0);
  gemm(stream, fcb, NOUTD, 0, 0, gT, NANSW, 0, 0, 1,
       d_out, NANSW, 0, 0, nullptr, 0, Bsz, NANSW, NOUTD, 1, 1, 0);
  k_logsoftmax<<<Bsz, 256, 0, stream>>>((float*)d_out, NANSW);
}
